// SNNTorchNet_14559939134164
// MI455X (gfx1250) — matrix operand order repaired and checked by reading, not compiled, not run on hardware
//
#include <hip/hip_runtime.h>
#include <hip/hip_bf16.h>

typedef __attribute__((ext_vector_type(16))) __bf16 v16bf;
typedef __attribute__((ext_vector_type(8)))  float  v8f;

#define T_STEPS 4000
#define NWG 8
#define THREADS 256
#define BETA 0.97530991202833262f   // exp(-0.25/10)

#define FRAG_USH 512   // one 16x32 (A) / 32x16 (B) bf16 fragment: 32 lanes * 16 elems
#define KT0 2          // layer0 K=64  -> 2 k-tiles
#define KT1 32         // layer1 K=1024
#define KT2 32         // layer2 K=1024

// ---------- helpers ----------
static __device__ __forceinline__ unsigned short f2bf(float f) {
  union { float f; unsigned u; } v; v.f = f;
  unsigned u = v.u;
  u = u + 0x7FFFu + ((u >> 16) & 1u);       // round-to-nearest-even
  return (unsigned short)(u >> 16);
}

// ---------- one-time pack kernels ----------
// Pack W[K][N] (fp32 row-major) -> bf16 WMMA B-fragments, with Dale ReLU clamp.
// B 32x16 bf16 layout: elem e -> k=base+(lane/16)*8+(e&7)+(e>>3)*16,
//                      n = ntile*16 + lane%16.
__global__ void pack_weights(const float* __restrict__ W, int K, int N, int NT,
                             unsigned short* __restrict__ dst) {
  int tid = blockIdx.x * blockDim.x + threadIdx.x;
  int KT = K >> 5;
  if (tid >= NT * KT * 32) return;
  int lane = tid & 31;
  int kt   = (tid >> 5) % KT;
  int nt   = (tid >> 5) / KT;
  int n    = nt * 16 + (lane & 15);
  int kb   = kt * 32 + ((lane >> 4) << 3);
  unsigned short* out = dst + (size_t)tid * 16;
  #pragma unroll
  for (int e = 0; e < 16; ++e) {
    float val = 0.f;
    if (n < N) {
      val = W[(size_t)(kb + (e & 7) + ((e >> 3) << 4)) * N + n];
      val = val > 0.f ? val : 0.f;          // max(W,0)
    }
    out[e] = f2bf(val);
  }
}

// Pack input_spikes [T][16][64] fp32 -> bf16 WMMA A-fragments per step.
// A 16x32 bf16 layout: m = lane%16; k = kt*32 + (lane/16)*8 + (e&7) + (e>>3)*16.
__global__ void pack_input(const float* __restrict__ in, unsigned short* __restrict__ dst) {
  int tid = blockIdx.x * blockDim.x + threadIdx.x;      // (t, kt, lane)
  if (tid >= T_STEPS * KT0 * 32) return;
  int lane = tid & 31;
  int kt   = (tid >> 5) & (KT0 - 1);
  int t    = tid >> 6;
  const float* row = in + ((size_t)t * 16 + (lane & 15)) * 64;
  int gh = (lane >> 4) << 3;
  unsigned short* out = dst + (size_t)tid * 16;
  #pragma unroll
  for (int e = 0; e < 16; ++e) {
    int k = kt * 32 + gh + (e & 7) + ((e >> 3) << 4);
    out[e] = f2bf(row[k]);
  }
}

// ---------- grid-wide phase barrier (persistent kernel, 8 WGs) ----------
static __device__ __forceinline__ void grid_barrier(unsigned* ctr, unsigned& target) {
  target += NWG;
  __threadfence();                 // release stores from all waves
  __syncthreads();
  if (threadIdx.x == 0) {
    atomicAdd(ctr, 1u);
    while (atomicAdd(ctr, 0u) < target) __builtin_amdgcn_s_sleep(1);
  }
  __syncthreads();
  __threadfence();                 // acquire
}

// ---------- persistent SNN kernel ----------
// 8 WGs x 256 threads (8 wave32 each). Wave (wg*8+wv) owns one 16-column tile
// of BOTH hidden layers. W1/W0 slices live in LDS (272KB, +32KB W2 on WG0).
// Membranes live in VGPRs as WMMA C/D fragments across all 4000 steps.
__launch_bounds__(THREADS, 1)
__global__ void snn_persistent(const unsigned short* __restrict__ inPacked,
                               const unsigned short* __restrict__ w0p,
                               const unsigned short* __restrict__ w1p,
                               const unsigned short* __restrict__ w2p,
                               unsigned short* __restrict__ sp0,  // 2 x 32KB (A-frag layout)
                               unsigned short* __restrict__ sp1,  // 2 x 32KB
                               const float* __restrict__ b0,
                               const float* __restrict__ b1,
                               const float* __restrict__ b2,
                               float* __restrict__ out,
                               unsigned* __restrict__ bar_ctr) {
  extern __shared__ unsigned short smem[];
  unsigned short* w1s = smem;                                   // 8*32*512 ush = 256KB
  unsigned short* w0s = w1s + 8 * KT1 * FRAG_USH;               // 8*2*512  ush = 16KB
  unsigned short* w2s = w0s + 8 * KT0 * FRAG_USH;               // 32*512   ush = 32KB

  const int tid  = threadIdx.x;
  const int wg   = blockIdx.x;
  const int lane = tid & 31;
  const int wv   = tid >> 5;

  // stage packed weight slices into LDS once
  {
    const uint4* s = (const uint4*)(w1p + (size_t)wg * 8 * KT1 * FRAG_USH);
    uint4* d = (uint4*)w1s;
    for (int i = tid; i < (8 * KT1 * FRAG_USH) / 8; i += THREADS) d[i] = s[i];
  }
  {
    const uint4* s = (const uint4*)(w0p + (size_t)wg * 8 * KT0 * FRAG_USH);
    uint4* d = (uint4*)w0s;
    for (int i = tid; i < (8 * KT0 * FRAG_USH) / 8; i += THREADS) d[i] = s[i];
  }
  if (wg == 0) {
    const uint4* s = (const uint4*)w2p;
    uint4* d = (uint4*)w2s;
    for (int i = tid; i < (KT2 * FRAG_USH) / 8; i += THREADS) d[i] = s[i];
  }
  __syncthreads();

  const int gnt  = wg * 8 + wv;            // global 16-col tile id (0..63)
  const int col  = gnt * 16 + (lane & 15); // this lane's output column
  const float bias0 = b0[col];
  const float bias1 = b1[col];
  const float bias2 = ((lane & 15) < 10) ? b2[lane & 15] : 0.f;

  // scatter indices: spike (m,col) -> consumer A-fragment position
  const int skt    = col >> 5;
  const int kp     = col & 31;
  const int hi     = kp >> 4;
  const int within = kp & 15;
  const int sg     = within >> 3;
  const int se     = (within & 7) + (hi << 3);

  float m0f[8], m1f[8], m2f[8];
  #pragma unroll
  for (int r = 0; r < 8; ++r) { m0f[r] = 0.f; m1f[r] = 0.f; m2f[r] = 0.f; }

  unsigned bar_target = 0;
  const int rbase = (lane >> 4) << 3;      // C/D frag: row = r + 8*(lane/16)

  for (int t = 0; t < T_STEPS; ++t) {
    unsigned short* sp0cur = sp0 + (size_t)(t & 1) * (KT1 * FRAG_USH);
    unsigned short* sp1cur = sp1 + (size_t)(t & 1) * (KT2 * FRAG_USH);

    // ---------- layer 0: [16x64] @ [64x1024] ----------
    {
      const unsigned short* aB = inPacked + (size_t)t * KT0 * FRAG_USH + lane * 16;
      const unsigned short* bB = w0s + wv * KT0 * FRAG_USH + lane * 16;
      v8f acc = {0.f,0.f,0.f,0.f,0.f,0.f,0.f,0.f};
      #pragma unroll
      for (int kt = 0; kt < KT0; ++kt) {
        v16bf a = *(const v16bf*)(aB + kt * FRAG_USH);
        v16bf b = *(const v16bf*)(bB + kt * FRAG_USH);
        acc = __builtin_amdgcn_wmma_f32_16x16x32_bf16(false, a, false, b,
                                                      (short)0, acc, false, false);
      }
      #pragma unroll
      for (int r = 0; r < 8; ++r) {
        float m = BETA * m0f[r] + acc[r] + bias0;
        int sp  = (m - 1.0f) >= 0.f;
        m0f[r]  = sp ? 0.f : m;
        int mrow = r + rbase;
        sp0cur[((size_t)skt * 32 + (mrow + (sg << 4))) * 16 + se] =
            sp ? (unsigned short)0x3F80 : (unsigned short)0;
      }
    }
    grid_barrier(bar_ctr, bar_target);

    // ---------- layer 1: [16x1024] @ [1024x1024] ----------
    {
      const unsigned short* aB = sp0cur + lane * 16;
      const unsigned short* bB = w1s + wv * KT1 * FRAG_USH + lane * 16;
      v8f acc = {0.f,0.f,0.f,0.f,0.f,0.f,0.f,0.f};
      #pragma unroll 4
      for (int kt = 0; kt < KT1; ++kt) {
        v16bf a = *(const v16bf*)(aB + kt * FRAG_USH);
        v16bf b = *(const v16bf*)(bB + kt * FRAG_USH);
        acc = __builtin_amdgcn_wmma_f32_16x16x32_bf16(false, a, false, b,
                                                      (short)0, acc, false, false);
      }
      #pragma unroll
      for (int r = 0; r < 8; ++r) {
        float m = BETA * m1f[r] + acc[r] + bias1;
        int sp  = (m - 1.0f) >= 0.f;
        m1f[r]  = sp ? 0.f : m;
        int mrow = r + rbase;
        sp1cur[((size_t)skt * 32 + (mrow + (sg << 4))) * 16 + se] =
            sp ? (unsigned short)0x3F80 : (unsigned short)0;
      }
    }
    grid_barrier(bar_ctr, bar_target);

    // ---------- layer 2: [16x1024] @ [1024x10] (WG0 wave0 only) ----------
    if (wg == 0 && wv == 0) {
      const unsigned short* aB = sp1cur + lane * 16;
      const unsigned short* bB = w2s + lane * 16;
      v8f acc = {0.f,0.f,0.f,0.f,0.f,0.f,0.f,0.f};
      #pragma unroll 4
      for (int kt = 0; kt < KT2; ++kt) {
        v16bf a = *(const v16bf*)(aB + kt * FRAG_USH);
        v16bf b = *(const v16bf*)(bB + kt * FRAG_USH);
        acc = __builtin_amdgcn_wmma_f32_16x16x32_bf16(false, a, false, b,
                                                      (short)0, acc, false, false);
      }
      #pragma unroll
      for (int r = 0; r < 8; ++r) {
        float m = BETA * m2f[r] + acc[r] + bias2;
        int sp  = (m - 1.0f) >= 0.f;
        m2f[r]  = sp ? 0.f : m;
        int n = lane & 15;
        if (n < 10)
          out[((size_t)t * 16 + (r + rbase)) * 10 + n] = sp ? 1.f : 0.f;
      }
    }
  }
}

// ---------- host launch ----------
extern "C" void kernel_launch(void* const* d_in, const int* in_sizes, int n_in,
                              void* d_out, int out_size, void* d_ws, size_t ws_size,
                              hipStream_t stream) {
  const float* input = (const float*)d_in[0];   // [4000,16,64]
  const float* W0 = (const float*)d_in[1];      // [64,1024]
  const float* b0 = (const float*)d_in[2];
  const float* W1 = (const float*)d_in[3];      // [1024,1024]
  const float* b1 = (const float*)d_in[4];
  const float* W2 = (const float*)d_in[5];      // [1024,10]
  const float* b2 = (const float*)d_in[6];
  float* out = (float*)d_out;

  char* ws = (char*)d_ws;
  size_t off = 0;
  auto alloc = [&](size_t bytes) -> char* {
    off = (off + 255) & ~(size_t)255;
    char* p = ws + off;
    off += bytes;
    return p;
  };
  unsigned short* inPacked = (unsigned short*)alloc((size_t)T_STEPS * KT0 * FRAG_USH * 2);
  unsigned short* w0p = (unsigned short*)alloc((size_t)64 * KT0 * FRAG_USH * 2);
  unsigned short* w1p = (unsigned short*)alloc((size_t)64 * KT1 * FRAG_USH * 2);
  unsigned short* w2p = (unsigned short*)alloc((size_t)1  * KT2 * FRAG_USH * 2);
  unsigned short* sp0 = (unsigned short*)alloc((size_t)2 * KT1 * FRAG_USH * 2);
  unsigned short* sp1 = (unsigned short*)alloc((size_t)2 * KT2 * FRAG_USH * 2);
  unsigned* ctr = (unsigned*)alloc(256);

  hipMemsetAsync(ctr, 0, sizeof(unsigned), stream);

  pack_input<<<(T_STEPS * KT0 * 32 + 255) / 256, 256, 0, stream>>>(input, inPacked);
  pack_weights<<<(64 * KT0 * 32 + 255) / 256, 256, 0, stream>>>(W0, 64,   1024, 64, w0p);
  pack_weights<<<(64 * KT1 * 32 + 255) / 256, 256, 0, stream>>>(W1, 1024, 1024, 64, w1p);
  pack_weights<<<(1  * KT2 * 32 + 255) / 256, 256, 0, stream>>>(W2, 1024, 10,   1,  w2p);

  size_t smem = (size_t)(8 * KT1 * FRAG_USH + 8 * KT0 * FRAG_USH + KT2 * FRAG_USH)
                * sizeof(unsigned short);   // 304 KB <= 320 KB WGP LDS
  snn_persistent<<<NWG, THREADS, smem, stream>>>(inPacked, w0p, w1p, w2p,
                                                 sp0, sp1, b0, b1, b2, out, ctr);
}
